// CrossFusionMamba_50362786512967
// MI455X (gfx1250) — compile-verified, matrix operand order checked by reading, not run on hardware
//
#include <hip/hip_runtime.h>
#include <cstddef>
#include <cstdint>

// ---------------------------------------------------------------------------
// CrossFusionMamba forward for MI455X (gfx1250, wave32).
// GEMMs: v_wmma_f32_16x16x32_bf16; A tile staged in LDS via the Tensor Data
// Mover (tensor_load_to_lds + s_wait_tensorcnt), shared by 4 waves/workgroup.
// Selective scan: 1 thread per (b, channel), 64-state vector in VGPRs.
// ---------------------------------------------------------------------------

#define B0     4
#define IMG    224
#define PATCH  16
#define Dm     256
#define NTOK   196
#define MROWS  (B0 * NTOK)   // 784
#define DIm    512
#define NST    64
#define DTRm   16
#define HEADS  4
#define DH     64
#define KPMAX  768

typedef unsigned short u16;
typedef __attribute__((ext_vector_type(16))) __bf16        v16bf;
typedef __attribute__((ext_vector_type(8)))  float         v8f;
typedef __attribute__((ext_vector_type(8)))  unsigned int  v8u;
typedef __attribute__((ext_vector_type(4)))  unsigned int  v4u;
typedef __attribute__((ext_vector_type(4)))  unsigned int  u32x4;
typedef __attribute__((ext_vector_type(8)))  int           i32x8;
typedef __attribute__((ext_vector_type(4)))  int           i32x4;

__device__ __forceinline__ u16 f2bf(float f) {
  unsigned int u = __builtin_bit_cast(unsigned int, f);
  u += 0x7fffu + ((u >> 16) & 1u);          // round-to-nearest-even
  return (u16)(u >> 16);
}
__device__ __forceinline__ float siluf(float x) { return x / (1.f + __expf(-x)); }

// ---------------------------------------------------------------------------
// WMMA GEMM: C[M,N] = A[Mp,Kp] @ B[Kp,N] (+bias[N]) (+resid[M,N])
// A: row-major bf16, Kp multiple of 32, zero-padded. Bt: N x Kp bf16,
// rows padded to a multiple of 64 and zero-filled.
// Workgroup = 4 waves covering a 16(M) x 64(N) strip; the 16 x Kp A tile is
// DMA'd to LDS once by the TDM and read back as ds_load_b128 fragments.
// A-fragment lane layout per CDNA5 ISA 7.12.2 (16-bit A 16x32).
// B-fragment: mirrored layout assumption (lane = N column, same K pattern).
// ---------------------------------------------------------------------------
__global__ void __launch_bounds__(128) wmma_gemm_k(
    const u16* __restrict__ A, const u16* __restrict__ Bt,
    float* __restrict__ C, const float* __restrict__ bias,
    const float* __restrict__ resid, int M, int N, int Kp, int ldc) {
  __shared__ __align__(16) u16 sA[16 * KPMAX];
  const int tile_m = blockIdx.x * 16;
  const int wave   = threadIdx.x >> 5;
  const int lane   = threadIdx.x & 31;
  const int tile_n = (blockIdx.y * 4 + wave) * 16;

  // ---- stage A tile (16 rows x Kp bf16) into LDS ----
#if __has_builtin(__builtin_amdgcn_tensor_load_to_lds)
  if (threadIdx.x < 32) {     // one TDM issue per workgroup (wave 0)
    unsigned long long ga = (unsigned long long)(const void*)(A + (size_t)tile_m * Kp);
    unsigned int lds = (unsigned int)(unsigned long long)(const void*)&sA[0];
    u32x4 g0 = {}; i32x8 g1 = {}; i32x4 g2 = {}; i32x4 g3 = {};
    g0[0] = 1u;                                           // count=1, user mode
    g0[1] = lds;                                          // lds_addr
    g0[2] = (unsigned int)(ga & 0xffffffffu);             // global_addr[31:0]
    g0[3] = (unsigned int)((ga >> 32) & 0x1ffffffu)       // global_addr[56:32]
            | (2u << 30);                                 // type=2 (image)
    g1[0] = (int)(1u << 16);                              // data_size=1 (2 bytes)
    g1[1] = (int)((unsigned)Kp << 16);                    // tensor_dim0[15:0]
    g1[2] = (int)((((unsigned)Kp >> 16) & 0xffffu)        // tensor_dim0[31:16]
            | (16u << 16));                               // tensor_dim1[15:0] = 16
    g1[3] = (int)((unsigned)Kp << 16);                    // tile_dim0 = Kp
    g1[4] = (int)16u;                                     // tile_dim1=16, tile_dim2=0
    g1[5] = (int)(unsigned)Kp;                            // tensor_dim0_stride[31:0]
    g1[6] = 0; g1[7] = 0;
#if defined(__clang_major__) && (__clang_major__ >= 23)
    i32x8 g4 = {};
    __builtin_amdgcn_tensor_load_to_lds(g0, g1, g2, g3, g4, 0);
#else
    __builtin_amdgcn_tensor_load_to_lds(g0, g1, g2, g3, 0);
#endif
#if __has_builtin(__builtin_amdgcn_s_wait_tensorcnt)
    __builtin_amdgcn_s_wait_tensorcnt(0);
#else
    asm volatile("s_wait_tensorcnt 0x0" ::: "memory");
#endif
  }
  __syncthreads();
#else
  // fallback: cooperative dword copy
  {
    const unsigned int* src = (const unsigned int*)(A + (size_t)tile_m * Kp);
    unsigned int* dst = (unsigned int*)sA;
    int n32 = 16 * (Kp >> 1);
    for (int i = threadIdx.x; i < n32; i += 128) dst[i] = src[i];
  }
  __syncthreads();
#endif

  const int half = lane >> 4;          // 0 or 1
  const int l15  = lane & 15;
  const int kb   = half * 8;
  const u16* sArow = sA + l15 * Kp;
  const u16* Brow  = Bt + (size_t)(tile_n + l15) * Kp;

  v8f acc = {};
  for (int k0 = 0; k0 < Kp; k0 += 32) {
    if (k0 + 32 < Kp)                   // gfx1250 global_prefetch_b8
      __builtin_prefetch(Brow + k0 + 32, 0, 0);
    // fragment pairs live at [kb, kb+8) and [16+kb, 16+kb+8): two b128 each
    v4u alo = *(const v4u*)(sArow + k0 + kb);
    v4u ahi = *(const v4u*)(sArow + k0 + kb + 16);
    v4u blo = *(const v4u*)(Brow  + k0 + kb);
    v4u bhi = *(const v4u*)(Brow  + k0 + kb + 16);
    struct Pair { v4u lo, hi; };
    Pair pa { alo, ahi }, pb { blo, bhi };
    v16bf a = __builtin_bit_cast(v16bf, pa);
    v16bf b = __builtin_bit_cast(v16bf, pb);
    acc = __builtin_amdgcn_wmma_f32_16x16x32_bf16(false, a, false, b,
                                                  (short)0, acc, false, false);
  }
  const int n = tile_n + l15;
#pragma unroll
  for (int r = 0; r < 8; ++r) {
    const int m = tile_m + r + half * 8;
    if (m < M && n < N) {
      float v = acc[r];
      if (bias)  v += bias[n];
      if (resid) v += resid[(size_t)m * ldc + n];
      C[(size_t)m * ldc + n] = v;
    }
  }
}

// f32 -> bf16 with zero padding (rowsPad x Kp), optional row stride srcLd.
__global__ void cvt_pad_bf16_k(const float* __restrict__ src, u16* __restrict__ dst,
                               int rows, int cols, int srcLd, int Kp, int rowsPad) {
  int idx = blockIdx.x * blockDim.x + threadIdx.x;
  int total = rowsPad * Kp;
  if (idx >= total) return;
  int r = idx / Kp, c = idx % Kp;
  float v = (r < rows && c < cols) ? src[(size_t)r * srcLd + c] : 0.f;
  dst[idx] = f2bf(v);
}

// f32 (K x N row-major) -> bf16 transposed (Np x Kp), zero padded.
__global__ void cvtT_pad_bf16_k(const float* __restrict__ src, u16* __restrict__ dst,
                                int K, int N, int Kp, int Np) {
  int idx = blockIdx.x * blockDim.x + threadIdx.x;
  int total = Np * Kp;
  if (idx >= total) return;
  int n = idx / Kp, k = idx % Kp;
  float v = (k < K && n < N) ? src[(size_t)k * N + n] : 0.f;
  dst[idx] = f2bf(v);
}

// concat [a | b] (784x256 each) -> bf16 784x512
__global__ void concat2_bf16_k(const float* __restrict__ a, const float* __restrict__ b,
                               u16* __restrict__ dst) {
  int idx = blockIdx.x * blockDim.x + threadIdx.x;
  if (idx >= MROWS * 2 * Dm) return;
  int r = idx / (2 * Dm), c = idx % (2 * Dm);
  float v = (c < Dm) ? a[(size_t)r * Dm + c] : b[(size_t)r * Dm + (c - Dm)];
  dst[idx] = f2bf(v);
}

// im2col for 16x16 patches, NCHW fp32 -> (784 x 768) bf16; k = c*256+py*16+px
__global__ void im2col_bf16_k(const float* __restrict__ img, u16* __restrict__ dst) {
  int idx = blockIdx.x * blockDim.x + threadIdx.x;
  const int total = MROWS * 768;
  if (idx >= total) return;
  int row = idx / 768, kk = idx % 768;
  int b = row / NTOK, tok = row % NTOK;
  int ty = tok / 14, tx = tok % 14;
  int c = kk / 256, rem = kk % 256, py = rem / 16, px = rem % 16;
  float v = img[(((size_t)(b * 3 + c)) * IMG + (ty * PATCH + py)) * IMG + (tx * PATCH + px)];
  dst[idx] = f2bf(v);
}

// pallor stage 1: xn = ((x+1)/2) / (sum_c + 1e-6)
__global__ void pallor_norm_k(const float* __restrict__ x, float* __restrict__ xn) {
  int idx = blockIdx.x * blockDim.x + threadIdx.x;
  const int HW = IMG * IMG;
  if (idx >= B0 * HW) return;
  int b = idx / HW, pix = idx % HW;
  float xp0 = (x[((size_t)(b * 3 + 0)) * HW + pix] + 1.f) * 0.5f;
  float xp1 = (x[((size_t)(b * 3 + 1)) * HW + pix] + 1.f) * 0.5f;
  float xp2 = (x[((size_t)(b * 3 + 2)) * HW + pix] + 1.f) * 0.5f;
  float inv = 1.f / (xp0 + xp1 + xp2 + 1e-6f);
  xn[((size_t)(b * 3 + 0)) * HW + pix] = xp0 * inv;
  xn[((size_t)(b * 3 + 1)) * HW + pix] = xp1 * inv;
  xn[((size_t)(b * 3 + 2)) * HW + pix] = xp2 * inv;
}

// deterministic per-(b,c) spatial mean: one block per channel
__global__ void chan_mean_k(const float* __restrict__ xn, float* __restrict__ cmean) {
  __shared__ float red[256];
  const int HW = IMG * IMG;
  int bc = blockIdx.x, tid = threadIdx.x;
  const float* p = xn + (size_t)bc * HW;
  float s = 0.f;
  for (int i = tid; i < HW; i += 256) s += p[i];
  red[tid] = s; __syncthreads();
  for (int st = 128; st > 0; st >>= 1) { if (tid < st) red[tid] += red[tid + st]; __syncthreads(); }
  if (tid == 0) cmean[bc] = red[0] * (1.f / (float)HW);
}

__global__ void pallor_sub_k(float* __restrict__ xn, const float* __restrict__ cmean) {
  int idx = blockIdx.x * blockDim.x + threadIdx.x;
  const int HW = IMG * IMG;
  if (idx >= B0 * 3 * HW) return;
  xn[idx] -= cmean[idx / HW];
}

// LayerNorm over last dim (256), one block (256 threads) per row.
__global__ void layernorm_k(const float* __restrict__ x, float* __restrict__ y,
                            const float* __restrict__ g, const float* __restrict__ b) {
  __shared__ float red[256];
  int row = blockIdx.x, tid = threadIdx.x;
  float v = x[(size_t)row * Dm + tid];
  red[tid] = v; __syncthreads();
  for (int st = 128; st > 0; st >>= 1) { if (tid < st) red[tid] += red[tid + st]; __syncthreads(); }
  float mean = red[0] * (1.f / Dm); __syncthreads();
  float d = v - mean;
  red[tid] = d * d; __syncthreads();
  for (int st = 128; st > 0; st >>= 1) { if (tid < st) red[tid] += red[tid + st]; __syncthreads(); }
  float var = red[0] * (1.f / Dm);
  y[(size_t)row * Dm + tid] = d * rsqrtf(var + 1e-5f) * g[tid] + b[tid];
}

__global__ void addpos_k(float* __restrict__ y, const float* __restrict__ pos) {
  int idx = blockIdx.x * blockDim.x + threadIdx.x;
  if (idx >= MROWS * Dm) return;
  int t = (idx / Dm) % NTOK, d = idx % Dm;
  y[idx] += pos[(size_t)t * Dm + d];
}

// depthwise causal conv (k=4) + bias + SiLU; xz is 784x1024 (xr = cols 0..511)
__global__ void conv_silu_k(const float* __restrict__ xz, const float* __restrict__ w,
                            const float* __restrict__ cb, float* __restrict__ xc) {
  int idx = blockIdx.x * blockDim.x + threadIdx.x;
  if (idx >= MROWS * DIm) return;
  int m = idx / DIm, c = idx % DIm;
  int b = m / NTOK, l = m % NTOK;
  float acc = cb[c];
#pragma unroll
  for (int k = 0; k < 4; ++k) {
    int li = l - 3 + k;
    if (li >= 0) acc += xz[((size_t)(b * NTOK + li)) * (2 * DIm) + c] * w[c * 4 + k];
  }
  xc[idx] = siluf(acc);
}

__global__ void softplus_k(float* __restrict__ x, int n) {
  int idx = blockIdx.x * blockDim.x + threadIdx.x;
  if (idx >= n) return;
  float v = x[idx];
  x[idx] = (v > 20.f) ? v : log1pf(__expf(v));
}

// selective scan: thread = (b, channel). 64 states in VGPRs.
__global__ void mamba_scan_k(const float* __restrict__ dt, const float* __restrict__ xc,
                             const float* __restrict__ dbc, const float* __restrict__ A_log,
                             const float* __restrict__ Dp, float* __restrict__ y) {
  int idx = blockIdx.x * blockDim.x + threadIdx.x;
  if (idx >= B0 * DIm) return;
  int b = idx / DIm, c = idx % DIm;
  float negA[NST], h[NST];
#pragma unroll
  for (int n = 0; n < NST; ++n) { negA[n] = -__expf(A_log[(size_t)c * NST + n]); h[n] = 0.f; }
  for (int l = 0; l < NTOK; ++l) {
    int row = b * NTOK + l;
    float dtv = dt[(size_t)row * DIm + c];
    float xcv = xc[(size_t)row * DIm + c];
    float du = dtv * xcv;
    const float* Bp = dbc + (size_t)row * 144 + DTRm;
    const float* Cp = Bp + NST;
    float acc = 0.f;
#pragma unroll
    for (int n = 0; n < NST; ++n) {
      h[n] = h[n] * __expf(dtv * negA[n]) + du * Bp[n];
      acc += h[n] * Cp[n];
    }
    y[(size_t)row * DIm + c] = acc + xcv * Dp[c];
  }
}

// out = bf16(y * silu(z)) where z = xz[:, 512:]
__global__ void gate_silu_bf16_k(const float* __restrict__ y, const float* __restrict__ xz,
                                 u16* __restrict__ out) {
  int idx = blockIdx.x * blockDim.x + threadIdx.x;
  if (idx >= MROWS * DIm) return;
  int m = idx / DIm, c = idx % DIm;
  float z = xz[(size_t)m * (2 * DIm) + DIm + c];
  out[idx] = f2bf(y[idx] * siluf(z));
}

// attention: block per (head, batch); K tile cached in LDS (196x64 f32 = 50KB)
__global__ void mha_attn_k(const float* __restrict__ q, const float* __restrict__ k,
                           const float* __restrict__ v, float* __restrict__ o) {
  __shared__ float Ks[NTOK * DH];
  __shared__ float sc[256];
  __shared__ float red[256];
  __shared__ float qr[DH];
  int h = blockIdx.x, b = blockIdx.y, tid = threadIdx.x;
  for (int i = tid; i < NTOK * DH; i += 256) {
    int r = i / DH, d = i % DH;
    Ks[i] = k[((size_t)(b * NTOK + r)) * Dm + h * DH + d];
  }
  __syncthreads();
  for (int qi = 0; qi < NTOK; ++qi) {
    if (tid < DH) qr[tid] = q[((size_t)(b * NTOK + qi)) * Dm + h * DH + tid];
    __syncthreads();
    float s = -1e30f;
    if (tid < NTOK) {
      float acc = 0.f;
#pragma unroll 8
      for (int d = 0; d < DH; ++d) acc += qr[d] * Ks[tid * DH + d];
      s = acc * 0.125f;                       // 1/sqrt(64)
    }
    sc[tid] = s; red[tid] = s; __syncthreads();
    for (int st = 128; st > 0; st >>= 1) { if (tid < st) red[tid] = fmaxf(red[tid], red[tid + st]); __syncthreads(); }
    float mx = red[0]; __syncthreads();
    float p = (tid < NTOK) ? __expf(sc[tid] - mx) : 0.f;
    sc[tid] = p; red[tid] = p; __syncthreads();
    for (int st = 128; st > 0; st >>= 1) { if (tid < st) red[tid] += red[tid + st]; __syncthreads(); }
    float inv = 1.f / red[0]; __syncthreads();
    if (tid < DH) {
      float acc = 0.f;
      for (int kk = 0; kk < NTOK; ++kk)
        acc += sc[kk] * v[((size_t)(b * NTOK + kk)) * Dm + h * DH + tid];
      o[((size_t)(b * NTOK + qi)) * Dm + h * DH + tid] = acc * inv;
    }
    __syncthreads();
  }
}

__global__ void gated_add_k(const float* __restrict__ base, const float* __restrict__ attn,
                            const float* __restrict__ gate, float* __restrict__ out) {
  int idx = blockIdx.x * blockDim.x + threadIdx.x;
  if (idx >= MROWS * Dm) return;
  float g = fminf(fmaxf(gate[0], 0.f), 1.f);
  out[idx] = base[idx] + g * attn[idx];
}

// heads from broadcast cls token; out[0..7]=logits, out[8..11]=hb
__global__ void heads_k(const float* __restrict__ cls,
                        const float* __restrict__ ch1w, const float* __restrict__ ch1b,
                        const float* __restrict__ ch2w, const float* __restrict__ ch2b,
                        const float* __restrict__ rh1w, const float* __restrict__ rh1b,
                        const float* __restrict__ rh2w, const float* __restrict__ rh2b,
                        float* __restrict__ out) {
  __shared__ float h1[128], h2[128];
  int tid = threadIdx.x;
  float a = ch1b[tid], r = rh1b[tid];
  for (int i = 0; i < Dm; ++i) {
    float cv = cls[i];
    a += cv * ch1w[i * 128 + tid];
    r += cv * rh1w[i * 128 + tid];
  }
  h1[tid] = 0.5f * a * (1.f + erff(a * 0.70710678f));
  h2[tid] = 0.5f * r * (1.f + erff(r * 0.70710678f));
  __syncthreads();
  if (tid == 0) {
    float l0 = ch2b[0], l1 = ch2b[1], hb = rh2b[0];
    for (int j = 0; j < 128; ++j) {
      l0 += h1[j] * ch2w[j * 2 + 0];
      l1 += h1[j] * ch2w[j * 2 + 1];
      hb += h2[j] * rh2w[j];
    }
    for (int bb = 0; bb < B0; ++bb) {
      out[bb * 2 + 0] = l0;
      out[bb * 2 + 1] = l1;
      out[8 + bb] = hb;
    }
  }
}

// ---------------------------------------------------------------------------
// Host orchestration
// ---------------------------------------------------------------------------
extern "C" void kernel_launch(void* const* d_in, const int* in_sizes, int n_in,
                              void* d_out, int out_size, void* d_ws, size_t ws_size,
                              hipStream_t stream) {
  (void)in_sizes; (void)n_in; (void)out_size; (void)ws_size;
  auto F = [&](int i) -> const float* { return (const float*)d_in[i]; };

  // jax tree leaves of params, dict keys alphabetical, lists in order.
  const float* X      = F(0);
  const float* ch1_b  = F(1);  const float* ch1_w = F(2);
  const float* ch2_b  = F(3);  const float* ch2_w = F(4);
  const float* cls    = F(5);
  struct AttnW { const float *bk, *bo, *bq, *bv, *wk, *wo, *wq, *wv; };
  AttnW awAB { F(6),  F(7),  F(8),  F(9),  F(10), F(11), F(12), F(13) };
  AttnW awBA { F(14), F(15), F(16), F(17), F(18), F(19), F(20), F(21) };
  const float* gate_a = F(22); const float* gate_b = F(23);
  const float* na_b = F(24); const float* na_g = F(25);
  const float* nb_b = F(26); const float* nb_g = F(27);
  const float* fn_b = F(28); const float* fn_g = F(29);
  const float* fp_b = F(30); const float* fp_w = F(31);
  const float* pal_b  = F(32);
  struct BlockW { const float *lb, *lg, *A_log, *Dp, *conv_b, *conv_w, *dt_bias, *dt_proj, *in_proj, *out_proj, *x_proj; };
  auto readBlock = [&](int base) -> BlockW {
    return BlockW{ F(base+0), F(base+1), F(base+2), F(base+3), F(base+4), F(base+5),
                   F(base+6), F(base+7), F(base+8), F(base+9), F(base+10) };
  };
  BlockW palBlocks[4] = { readBlock(33), readBlock(44), readBlock(55), readBlock(66) };
  const float* pal_g = F(77); const float* pal_pb = F(78); const float* pal_pw = F(79);
  const float* pos_pallor = F(80); const float* pos_rgb = F(81);
  const float* ref_b = F(82);
  BlockW refBlocks[2] = { readBlock(83), readBlock(94) };
  const float* ref_g = F(105);
  const float* rgb_b = F(106);
  BlockW rgbBlocks[4] = { readBlock(107), readBlock(118), readBlock(129), readBlock(140) };
  const float* rgb_g = F(151); const float* rgb_pb = F(152); const float* rgb_pw = F(153);
  const float* rh1_b = F(154); const float* rh1_w = F(155);
  const float* rh2_b = F(156); const float* rh2_w = F(157);
  float* OUT = (float*)d_out;

  // workspace bump allocator
  char* base = (char*)d_ws;
  size_t off = 0;
  auto allocB = [&](size_t bytes) -> void* {
    void* p = base + off;
    off = (off + bytes + 255) & ~(size_t)255;
    return p;
  };
  auto allocF = [&](size_t n) -> float* { return (float*)allocB(n * 4); };
  auto allocH = [&](size_t n) -> u16*   { return (u16*)allocB(n * 2); };

  auto cvtPad = [&](const float* src, u16* dst, int rows, int cols, int srcLd, int Kp, int rowsPad) {
    int total = rowsPad * Kp;
    cvt_pad_bf16_k<<<(total + 255) / 256, 256, 0, stream>>>(src, dst, rows, cols, srcLd, Kp, rowsPad);
  };
  auto cvtT = [&](const float* src, u16* dst, int K, int N, int Kp, int Np) {
    int total = Np * Kp;
    cvtT_pad_bf16_k<<<(total + 255) / 256, 256, 0, stream>>>(src, dst, K, N, Kp, Np);
  };
  auto gemm = [&](const u16* A, const u16* Bt, float* C, const float* bias,
                  const float* resid, int M, int N, int Kp, int ldc) {
    dim3 g(M / 16, (N + 63) / 64);         // 4 waves/block cover 64 N
    wmma_gemm_k<<<g, 128, 0, stream>>>(A, Bt, C, bias, resid, M, N, Kp, ldc);
  };
  auto lnorm = [&](const float* x, float* y, const float* g, const float* b) {
    layernorm_k<<<MROWS, 256, 0, stream>>>(x, y, g, b);
  };

  // ---- bf16 weight prep (every launch; deterministic) --------------------
  u16* pwT_rgb = allocH((size_t)Dm * 768);
  u16* pwT_pal = allocH((size_t)Dm * 768);
  // pw is (256, 768) row-major == (N x K) already
  cvtPad(rgb_pw, pwT_rgb, Dm, 768, 768, 768, Dm);
  cvtPad(pal_pw, pwT_pal, Dm, 768, 768, 768, Dm);

  struct MBdev { u16 *inT, *xT, *dtT, *outT; };
  auto prepMB = [&](const BlockW& w) -> MBdev {
    MBdev m;
    m.inT  = allocH((size_t)1024 * 256);
    m.xT   = allocH((size_t)192 * 512);    // N=144 padded to 192 (mult of 64)
    m.dtT  = allocH((size_t)512 * 32);
    m.outT = allocH((size_t)256 * 512);
    cvtT(w.in_proj,  m.inT, 256, 1024, 256, 1024);
    cvtT(w.x_proj,   m.xT,  512, 144,  512, 192);
    cvtT(w.dt_proj,  m.dtT, 16,  512,  32,  512);
    cvtT(w.out_proj, m.outT,512, 256,  512, 256);
    return m;
  };
  MBdev rgbMB[4], palMB[4], refMB[2];
  for (int j = 0; j < 4; ++j) rgbMB[j] = prepMB(rgbBlocks[j]);
  for (int j = 0; j < 4; ++j) palMB[j] = prepMB(palBlocks[j]);
  for (int j = 0; j < 2; ++j) refMB[j] = prepMB(refBlocks[j]);

  struct AttnT { u16 *wqT, *wkT, *wvT, *woT; };
  auto prepAttn = [&](const AttnW& w) -> AttnT {
    AttnT t;
    t.wqT = allocH((size_t)Dm * Dm); t.wkT = allocH((size_t)Dm * Dm);
    t.wvT = allocH((size_t)Dm * Dm); t.woT = allocH((size_t)Dm * Dm);
    cvtT(w.wq, t.wqT, Dm, Dm, Dm, Dm); cvtT(w.wk, t.wkT, Dm, Dm, Dm, Dm);
    cvtT(w.wv, t.wvT, Dm, Dm, Dm, Dm); cvtT(w.wo, t.woT, Dm, Dm, Dm, Dm);
    return t;
  };
  AttnT atAB = prepAttn(awAB), atBA = prepAttn(awBA);
  u16* fpT = allocH((size_t)Dm * 512);
  cvtT(fp_w, fpT, 512, Dm, 512, Dm);

  // ---- activation scratch ------------------------------------------------
  float* pal    = allocF((size_t)B0 * 3 * IMG * IMG);
  float* cmean  = allocF(64);
  u16*   Aim    = allocH((size_t)MROWS * 768);
  float* bufA   = allocF((size_t)MROWS * Dm);
  float* bufB   = allocF((size_t)MROWS * Dm);
  float* tr     = allocF((size_t)MROWS * Dm);
  float* tp     = allocF((size_t)MROWS * Dm);
  float* xln    = allocF((size_t)MROWS * Dm);
  u16*   xln16  = allocH((size_t)MROWS * Dm);
  float* xz     = allocF((size_t)MROWS * 2 * DIm);
  float* xc     = allocF((size_t)MROWS * DIm);
  u16*   xc16   = allocH((size_t)MROWS * DIm);
  float* dbc    = allocF((size_t)MROWS * 144);
  u16*   dt16   = allocH((size_t)MROWS * 32);
  float* dtb    = allocF((size_t)MROWS * DIm);
  float* ybuf   = allocF((size_t)MROWS * DIm);
  u16*   yg16   = allocH((size_t)MROWS * DIm);
  u16*   tr16   = allocH((size_t)MROWS * Dm);
  u16*   tp16   = allocH((size_t)MROWS * Dm);
  float* qb     = allocF((size_t)MROWS * Dm);
  float* kb     = allocF((size_t)MROWS * Dm);
  float* vb     = allocF((size_t)MROWS * Dm);
  float* ob     = allocF((size_t)MROWS * Dm);
  u16*   ob16   = allocH((size_t)MROWS * Dm);
  float* attno  = allocF((size_t)MROWS * Dm);
  float* tmpf   = allocF((size_t)MROWS * Dm);
  float* abuf   = allocF((size_t)MROWS * Dm);
  float* bbuf   = allocF((size_t)MROWS * Dm);
  u16*   cat16  = allocH((size_t)MROWS * 2 * Dm);

  // ---- mamba block runner ------------------------------------------------
  auto runMB = [&](const BlockW& w, const MBdev& m, const float* xin, float* xout) {
    lnorm(xin, xln, w.lg, w.lb);
    cvtPad(xln, xln16, MROWS, Dm, Dm, Dm, MROWS);
    gemm(xln16, m.inT, xz, nullptr, nullptr, MROWS, 2 * DIm, Dm, 2 * DIm);
    conv_silu_k<<<(MROWS * DIm + 255) / 256, 256, 0, stream>>>(xz, w.conv_w, w.conv_b, xc);
    cvtPad(xc, xc16, MROWS, DIm, DIm, DIm, MROWS);
    gemm(xc16, m.xT, dbc, nullptr, nullptr, MROWS, 144, DIm, 144);
    cvtPad(dbc, dt16, MROWS, 16, 144, 32, MROWS);
    gemm(dt16, m.dtT, dtb, w.dt_bias, nullptr, MROWS, DIm, 32, DIm);
    softplus_k<<<(MROWS * DIm + 255) / 256, 256, 0, stream>>>(dtb, MROWS * DIm);
    mamba_scan_k<<<(B0 * DIm + 255) / 256, 256, 0, stream>>>(dtb, xc, dbc, w.A_log, w.Dp, ybuf);
    gate_silu_bf16_k<<<(MROWS * DIm + 255) / 256, 256, 0, stream>>>(ybuf, xz, yg16);
    gemm(yg16, m.outT, xout, nullptr, xin, MROWS, Dm, DIm, Dm);
  };

  auto runBranch = [&](const float* img, const float* pb, u16* pwT,
                       const BlockW* blocks, const MBdev* mbs,
                       const float* g, const float* b, float* dst) {
    im2col_bf16_k<<<(MROWS * 768 + 255) / 256, 256, 0, stream>>>(img, Aim);
    gemm(Aim, pwT, bufA, pb, nullptr, MROWS, Dm, 768, Dm);
    float* cur = bufA; float* nxt = bufB;
    for (int j = 0; j < 4; ++j) { runMB(blocks[j], mbs[j], cur, nxt); float* t = cur; cur = nxt; nxt = t; }
    lnorm(cur, dst, g, b);
  };

  auto runMHA = [&](const u16* q16, const u16* kv16, const AttnT& t, const AttnW& w, float* outb) {
    gemm(q16,  t.wqT, qb, w.bq, nullptr, MROWS, Dm, Dm, Dm);
    gemm(kv16, t.wkT, kb, w.bk, nullptr, MROWS, Dm, Dm, Dm);
    gemm(kv16, t.wvT, vb, w.bv, nullptr, MROWS, Dm, Dm, Dm);
    mha_attn_k<<<dim3(HEADS, B0), 256, 0, stream>>>(qb, kb, vb, ob);
    cvtPad(ob, ob16, MROWS, Dm, Dm, Dm, MROWS);
    gemm(ob16, t.woT, outb, w.bo, nullptr, MROWS, Dm, Dm, Dm);
  };

  // ---- pallor preprocessing ---------------------------------------------
  pallor_norm_k<<<(B0 * IMG * IMG + 255) / 256, 256, 0, stream>>>(X, pal);
  chan_mean_k<<<B0 * 3, 256, 0, stream>>>(pal, cmean);
  pallor_sub_k<<<(B0 * 3 * IMG * IMG + 255) / 256, 256, 0, stream>>>(pal, cmean);

  // ---- branches ----------------------------------------------------------
  runBranch(X,   rgb_pb, pwT_rgb, rgbBlocks, rgbMB, rgb_g, rgb_b, tr);
  addpos_k<<<(MROWS * Dm + 255) / 256, 256, 0, stream>>>(tr, pos_rgb);
  runBranch(pal, pal_pb, pwT_pal, palBlocks, palMB, pal_g, pal_b, tp);
  addpos_k<<<(MROWS * Dm + 255) / 256, 256, 0, stream>>>(tp, pos_pallor);

  // ---- cross attention ---------------------------------------------------
  cvtPad(tr, tr16, MROWS, Dm, Dm, Dm, MROWS);
  cvtPad(tp, tp16, MROWS, Dm, Dm, Dm, MROWS);
  runMHA(tr16, tp16, atAB, awAB, attno);
  gated_add_k<<<(MROWS * Dm + 255) / 256, 256, 0, stream>>>(tr, attno, gate_a, tmpf);
  lnorm(tmpf, abuf, na_g, na_b);
  runMHA(tp16, tr16, atBA, awBA, attno);
  gated_add_k<<<(MROWS * Dm + 255) / 256, 256, 0, stream>>>(tp, attno, gate_b, tmpf);
  lnorm(tmpf, bbuf, nb_g, nb_b);

  // ---- fusion + refinement ----------------------------------------------
  concat2_bf16_k<<<(MROWS * 2 * Dm + 255) / 256, 256, 0, stream>>>(abuf, bbuf, cat16);
  gemm(cat16, fpT, bufA, fp_b, nullptr, MROWS, Dm, 2 * Dm, Dm);
  lnorm(bufA, bufB, fn_g, fn_b);
  runMB(refBlocks[0], refMB[0], bufB, bufA);
  runMB(refBlocks[1], refMB[1], bufA, bufB);
  lnorm(bufB, OUT + 12, ref_g, ref_b);      // fused -> d_out[12:]

  // ---- heads -------------------------------------------------------------
  heads_k<<<1, 128, 0, stream>>>(cls, ch1_w, ch1_b, ch2_w, ch2_b,
                                 rh1_w, rh1_b, rh2_w, rh2_b, OUT);
}
